// LinkPredictionMPNN_73289321939189
// MI455X (gfx1250) — matrix operand order repaired and checked by reading, not compiled, not run on hardware
//
#include <hip/hip_runtime.h>
#include <cstdint>

// Problem constants (match reference)
#define NN   50000
#define EE   800000
#define INF_ 128
#define HH   64
#define LL   2
#define KK   5
#define EPSL 1e-3f
// Node rows padded to the 128-row GEMM block tile: stores are unconditional.
#define NP   (((NN + 127) / 128) * 128)   // 50048

typedef __attribute__((ext_vector_type(16))) _Float16 v16h;
typedef __attribute__((ext_vector_type(8)))  float    v8f;
typedef __attribute__((ext_vector_type(4)))  float    v4f;

__device__ __forceinline__ v8f wmma_f16(v16h a, v16h b, v8f c) {
  return __builtin_amdgcn_wmma_f32_16x16x32_f16(false, a, false, b, (short)0,
                                                c, false, false);
}

// ---------------------------------------------------------------------------
// Pack a row-major fp32 weight [K x 64] into the per-lane WMMA B-fragment
// layout for v_wmma_f32_16x16x32_f16:
//   tile (kt, nt): lane l holds column n = nt*16 + (l&15),
//   halves e=0..15 hold K = kt*32 + (e>>3)*16 + (l>>4)*8 + (e&7).
// Output index = ((kt*4 + nt)*32 + l)*16 + e  == tid by construction.
// ---------------------------------------------------------------------------
__global__ void pack_w_kernel(const float* __restrict__ W,
                              _Float16* __restrict__ out, int ktiles) {
  int tid = blockIdx.x * blockDim.x + threadIdx.x;
  if (tid >= ktiles * 2048) return;
  int j  = tid & 15;
  int l  = (tid >> 4) & 31;
  int nt = (tid >> 9) & 3;
  int kt = tid >> 11;
  int n = nt * 16 + (l & 15);
  int k = kt * 32 + ((j >> 3) << 4) + ((l >> 4) << 3) + (j & 7);
  out[tid] = (_Float16)W[k * HH + n];
}

__global__ void zero_f32_kernel(float* __restrict__ p, int n) {
  int i = blockIdx.x * blockDim.x + threadIdx.x;
  if (i < n) p[i] = 0.0f;
}

// ---------------------------------------------------------------------------
// Load all A fragments (16 rows x KDIM, fp32 -> fp16) for one wave tile.
// ISA 16-bit A 16x32 layout: halves 0..7 -> K=koff.., halves 8..15 -> K=16+koff..
// ---------------------------------------------------------------------------
template <int KT>
__device__ __forceinline__ void load_a_frags(const float* __restrict__ xr,
                                             v16h (&a)[KT]) {
#pragma unroll
  for (int kt = 0; kt < KT; ++kt) {
#pragma unroll
    for (int i = 0; i < 8; ++i) {
      a[kt][i]     = (_Float16)xr[kt * 32 + i];
      a[kt][i + 8] = (_Float16)xr[kt * 32 + 16 + i];
    }
  }
}

// ---------------------------------------------------------------------------
// Y[NP x 64] = X[nrows x KDIM] @ Wpacked (+ bias).  One wave = 16 rows.
// A fragments hoisted into registers; Y is row-padded so stores need no guard.
// ---------------------------------------------------------------------------
template <int KDIM>
__global__ __launch_bounds__(256) void gemm_wmma_kernel(
    const float* __restrict__ X, const _Float16* __restrict__ Wp,
    const float* __restrict__ bias, float* __restrict__ Y, int nrows) {
  constexpr int KT = KDIM / 32;
  const int lane = threadIdx.x & 31;
  const int wave = threadIdx.x >> 5;
  const int m0   = blockIdx.x * 128 + wave * 16;
  const int row  = m0 + (lane & 15);
  const int mr   = row < nrows ? row : nrows - 1;  // clamp A loads, EXEC full
  const int koff = (lane >> 4) << 3;               // 0 or 8
  const int mhi  = (lane >> 4) << 3;               // 0 or 8 (D-row offset)
  const int nlo  = lane & 15;

  v16h a[KT];
  load_a_frags<KT>(X + (size_t)mr * KDIM + koff, a);

#pragma unroll
  for (int nt = 0; nt < 4; ++nt) {
    v8f acc = {};
#pragma unroll
    for (int kt = 0; kt < KT; ++kt) {
      v16h b = *(const v16h*)(Wp + (((size_t)(kt * 4 + nt) * 32 + lane) << 4));
      acc = wmma_f16(a[kt], b, acc);
    }
    const int n  = nt * 16 + nlo;
    const float bv = bias ? bias[n] : 0.0f;
    float* yp = Y + (size_t)(m0 + mhi) * HH + n;   // padded: no bounds checks
#pragma unroll
    for (int v = 0; v < 8; ++v) yp[(size_t)v * HH] = acc[v] + bv;
  }
}

// ---------------------------------------------------------------------------
// Fused dual GEMM for the conv layers: YS = X@Ws, YD = X@Wd + biasD.
// X is the padded h buffer -> no clamps/guards anywhere; 16 WMMA per tile.
// ---------------------------------------------------------------------------
__global__ __launch_bounds__(256) void dual_gemm_wmma_kernel(
    const float* __restrict__ X, const _Float16* __restrict__ WpS,
    const _Float16* __restrict__ WpD, const float* __restrict__ biasD,
    float* __restrict__ YS, float* __restrict__ YD) {
  constexpr int KT = HH / 32;  // 2
  const int lane = threadIdx.x & 31;
  const int wave = threadIdx.x >> 5;
  const int m0   = blockIdx.x * 128 + wave * 16;
  const int row  = m0 + (lane & 15);
  const int koff = (lane >> 4) << 3;
  const int mhi  = (lane >> 4) << 3;
  const int nlo  = lane & 15;

  v16h a[KT];
  load_a_frags<KT>(X + (size_t)row * HH + koff, a);

#pragma unroll
  for (int nt = 0; nt < 4; ++nt) {
    v8f accS = {};
    v8f accD = {};
#pragma unroll
    for (int kt = 0; kt < KT; ++kt) {
      const size_t fi = ((size_t)(kt * 4 + nt) * 32 + lane) << 4;
      v16h bS = *(const v16h*)(WpS + fi);
      v16h bD = *(const v16h*)(WpD + fi);
      accS = wmma_f16(a[kt], bS, accS);
      accD = wmma_f16(a[kt], bD, accD);
    }
    const int n  = nt * 16 + nlo;
    const float bv = biasD[n];
    float* ysp = YS + (size_t)(m0 + mhi) * HH + n;
    float* ydp = YD + (size_t)(m0 + mhi) * HH + n;
#pragma unroll
    for (int v = 0; v < 8; ++v) {
      ysp[(size_t)v * HH] = accS[v];
      ydp[(size_t)v * HH] = accD[v] + bv;
    }
  }
}

__global__ void degree_kernel(const int* __restrict__ dst, int* __restrict__ deg) {
  int e = blockIdx.x * blockDim.x + threadIdx.x;
  if (e < EE) atomicAdd(&deg[dst[e]], 1);
}

// segment_sum of hs[src[e]] into agg[dst[e]] : 64 lanes per edge,
// one native global_atomic_add_f32 per element (L2-resident rows).
__global__ __launch_bounds__(256) void scatter_kernel(
    const float* __restrict__ hs, const int* __restrict__ src,
    const int* __restrict__ dst, float* __restrict__ agg) {
  long long tid = (long long)blockIdx.x * 256 + threadIdx.x;
  int e = (int)(tid >> 6);
  int c = (int)(tid & 63);
  int s = src[e];
  int d = dst[e];
  if (c == 0) {  // hide gather latency: global_prefetch_b8 a future row
    int ep = e + 2048;
    if (ep < EE) __builtin_prefetch(&hs[(size_t)src[ep] * HH], 0, 1);
  }
  float v = hs[(size_t)s * HH + c];
  __hip_atomic_fetch_add(&agg[(size_t)d * HH + c], v,
                         __ATOMIC_RELAXED, __HIP_MEMORY_SCOPE_AGENT);
}

// h = relu(BN(sigmoid(agg + deg*hdb) + softplus(h)))   [elementwise, N*64]
__global__ __launch_bounds__(256) void combine_kernel(
    float* __restrict__ h, const float* __restrict__ agg,
    const float* __restrict__ hdb, const int* __restrict__ deg,
    const float* __restrict__ gamma, const float* __restrict__ beta,
    const float* __restrict__ mean, const float* __restrict__ var) {
  int tid = blockIdx.x * 256 + threadIdx.x;   // exactly N*64 threads
  int n = tid >> 6, c = tid & 63;
  float a   = agg[tid] + (float)deg[n] * hdb[tid];
  float sig = 1.0f / (1.0f + __expf(-a));
  float hv  = h[tid];
  float sp  = fmaxf(hv, 0.0f) + log1pf(__expf(-fabsf(hv)));
  float t   = sig + sp;
  t = gamma[c] * (t - mean[c]) * rsqrtf(var[c] + EPSL) + beta[c];
  h[tid] = fmaxf(t, 0.0f);
}

// 16 lanes per edge, float4 loads, shuffle reduction within the half-wave.
__global__ __launch_bounds__(256) void pos_score_kernel(
    const float* __restrict__ h, const int* __restrict__ src,
    const int* __restrict__ dst, const float* __restrict__ wrel,
    float* __restrict__ out) {
  int tid = blockIdx.x * 256 + threadIdx.x;
  int e = tid >> 4, j = tid & 15;
  int s = src[e], d = dst[e];
  v4f a = *(const v4f*)(h + (size_t)s * HH + j * 4);
  v4f b = *(const v4f*)(h + (size_t)d * HH + j * 4);
  v4f w = *(const v4f*)(wrel + j * 4);
  float sum = a.x * b.x * w.x + a.y * b.y * w.y + a.z * b.z * w.z + a.w * b.w * w.w;
  sum += __shfl_xor(sum, 8, 16);
  sum += __shfl_xor(sum, 4, 16);
  sum += __shfl_xor(sum, 2, 16);
  sum += __shfl_xor(sum, 1, 16);
  if (j == 0) out[e] = sum;
}

__global__ __launch_bounds__(256) void neg_score_kernel(
    const float* __restrict__ h, const int* __restrict__ src,
    const int* __restrict__ ndst, const float* __restrict__ wrel,
    float* __restrict__ out) {
  int tid = blockIdx.x * 256 + threadIdx.x;
  int e = tid >> 4, j = tid & 15;   // e in [0, E*K)
  int s = src[e / KK];              // neg_src = repeat(src, K)
  int d = ndst[e];
  v4f a = *(const v4f*)(h + (size_t)s * HH + j * 4);
  v4f b = *(const v4f*)(h + (size_t)d * HH + j * 4);
  v4f w = *(const v4f*)(wrel + j * 4);
  float sum = a.x * b.x * w.x + a.y * b.y * w.y + a.z * b.z * w.z + a.w * b.w * w.w;
  sum += __shfl_xor(sum, 8, 16);
  sum += __shfl_xor(sum, 4, 16);
  sum += __shfl_xor(sum, 2, 16);
  sum += __shfl_xor(sum, 1, 16);
  if (j == 0) out[e] = sum;
}

extern "C" void kernel_launch(void* const* d_in, const int* in_sizes, int n_in,
                              void* d_out, int out_size, void* d_ws, size_t ws_size,
                              hipStream_t stream) {
  (void)in_sizes; (void)n_in; (void)out_size; (void)ws_size;
  const float* node_feat = (const float*)d_in[0];
  const float* emb_w     = (const float*)d_in[1];
  const float* emb_b     = (const float*)d_in[2];
  const float* conv_w    = (const float*)d_in[3];
  const float* conv_b    = (const float*)d_in[4];
  const float* bn_gamma  = (const float*)d_in[5];
  const float* bn_beta   = (const float*)d_in[6];
  const float* bn_mean   = (const float*)d_in[7];
  const float* bn_var    = (const float*)d_in[8];
  const float* w_rel     = (const float*)d_in[9];
  const int*   src       = (const int*)d_in[10];
  const int*   dst       = (const int*)d_in[11];
  const int*   neg_dst   = (const int*)d_in[12];
  float* out = (float*)d_out;

  // Workspace layout (padded to NP rows; all arrays fit in MI455X's 192MB L2)
  const size_t NHP = (size_t)NP * HH;     // padded rows for guard-free stores
  const size_t NH  = (size_t)NN * HH;
  float* h   = (float*)d_ws;
  float* hs  = h + NHP;
  float* hdb = hs + NHP;
  float* agg = hdb + NHP;
  int*   deg = (int*)(agg + NHP);
  _Float16* pw = (_Float16*)(((uintptr_t)(deg + NN) + 63) & ~(uintptr_t)63);
  _Float16* pw_emb  = pw;                     // 128x64 packed
  _Float16* pw_conv = pw + INF_ * HH;         // 4 x (64x64) packed

  // Pack weights into WMMA B-fragment layout (tiny, once per launch).
  pack_w_kernel<<<32, 256, 0, stream>>>(emb_w, pw_emb, 4);
  for (int l = 0; l < LL; ++l) {
    const float* base = conv_w + (size_t)l * 2 * HH * HH;
    pack_w_kernel<<<16, 256, 0, stream>>>(base,           pw_conv + (2 * l) * HH * HH, 2);
    pack_w_kernel<<<16, 256, 0, stream>>>(base + HH * HH, pw_conv + (2 * l + 1) * HH * HH, 2);
  }

  // In-degrees (reused by both layers).
  zero_f32_kernel<<<(NN + 255) / 256, 256, 0, stream>>>((float*)deg, NN);
  degree_kernel<<<(EE + 255) / 256, 256, 0, stream>>>(dst, deg);

  // h = node_feat @ emb_w + emb_b   (WMMA, K=128; writes all NP padded rows)
  gemm_wmma_kernel<INF_><<<NP / 128, 256, 0, stream>>>(
      node_feat, pw_emb, emb_b, h, NN);

  for (int l = 0; l < LL; ++l) {
    // hs = h @ w_s ; hdb = h @ w_d + conv_b  (fused dual WMMA GEMM, K=64)
    dual_gemm_wmma_kernel<<<NP / 128, 256, 0, stream>>>(
        h, pw_conv + (2 * l) * HH * HH, pw_conv + (2 * l + 1) * HH * HH,
        conv_b + l * HH, hs, hdb);
    // agg = segment_sum(hs[src], dst)  -- the deg*hdb part is folded in combine
    zero_f32_kernel<<<(int)(NH / 256), 256, 0, stream>>>(agg, (int)NH);
    scatter_kernel<<<(int)((size_t)EE * HH / 256), 256, 0, stream>>>(hs, src, dst, agg);
    combine_kernel<<<(int)(NH / 256), 256, 0, stream>>>(
        h, agg, hdb, deg,
        bn_gamma + l * HH, bn_beta + l * HH, bn_mean + l * HH, bn_var + l * HH);
  }

  pos_score_kernel<<<(int)((size_t)EE * 16 / 256), 256, 0, stream>>>(h, src, dst, w_rel, out);
  neg_score_kernel<<<(int)((size_t)EE * KK * 16 / 256), 256, 0, stream>>>(h, src, neg_dst, w_rel, out + EE);
}
